// OreoTypeHeadsMLP_7112465842283
// MI455X (gfx1250) — compile-verified, matrix operand order checked by reading, not run on hardware
//
#include <hip/hip_runtime.h>
#include <math.h>

typedef __attribute__((ext_vector_type(16))) _Float16 v16h;
typedef __attribute__((ext_vector_type(8)))  _Float16 v8h;
typedef __attribute__((ext_vector_type(8)))  float    v8f;
typedef __attribute__((ext_vector_type(4)))  float    v4f;

#define B_ROWS  131072
#define IN_DIM  1024
#define H_DIM   384
#define L_DIM   128
#define K_SLOTS 32
#define TAU     0.7f

// packed WMMA B-fragment regions inside d_ws (offsets in _Float16 elements)
#define W1P_OFF 0
#define W1P_N   (IN_DIM * H_DIM)       // 393216
#define W2P_OFF (W1P_OFF + W1P_N)
#define W2P_N   (H_DIM * L_DIM)        // 49152
#define W3P_OFF (W2P_OFF + W2P_N)
#define W3P_N   (2 * L_DIM * H_DIM)    // 98304
#define MKP_OFF (W3P_OFF + W3P_N)
#define MKP_N   (L_DIM * K_SLOTS)      // 4096

__device__ __forceinline__ float gelu_exact(float x) {
  return 0.5f * x * (1.0f + erff(x * 0.70710678118654752f));
}

// Pack a [Kdim x N] f32 matrix (row-major, leading dim src_ld; or transposed
// source W[n*src_ld + k]) into per-fragment WMMA-B layout:
//   out[((ntile*ksteps + kstep)*32 + lane)*16 + slot]
// lane in [0,32): column n = ntile*16 + (lane&15); lanes>=16 take k-offset +8.
// slot order per lane: k = kstep*32 + hi8 + (slot<8 ? slot : slot+8)
__global__ void pack_b_kernel(const float* __restrict__ W, _Float16* __restrict__ out,
                              int Kdim, int N, int src_ld, int transposed) {
  int tid = blockIdx.x * blockDim.x + threadIdx.x;
  if (tid >= Kdim * N) return;
  int ksteps = Kdim >> 5;
  int slot = tid & 15;
  int lane = (tid >> 4) & 31;
  int frag = tid >> 9;
  int kstep = frag % ksteps;
  int ntile = frag / ksteps;
  int n   = (ntile << 4) + (lane & 15);
  int hi8 = (lane >> 4) << 3;
  int k   = (kstep << 5) + hi8 + ((slot < 8) ? slot : (slot + 8));
  float v = transposed ? W[n * src_ld + k] : W[k * src_ld + n];
  out[tid] = (_Float16)v;
}

#define WMMA(A, Bf, C) \
  __builtin_amdgcn_wmma_f32_16x16x32_f16(false, (A), false, (Bf), (short)0, (C), false, false)

__global__ __launch_bounds__(256) void fused_forward(
    const float* __restrict__ x,
    const _Float16* __restrict__ w1p,
    const _Float16* __restrict__ w2p,
    const _Float16* __restrict__ w3p,
    const _Float16* __restrict__ mkp,
    const float* __restrict__ b1,
    const float* __restrict__ b2,
    const float* __restrict__ b3,
    const float* __restrict__ b4,
    const float* __restrict__ mv,
    const float* __restrict__ w4,
    float* __restrict__ out)
{
  __shared__ __attribute__((aligned(16))) _Float16 sh_h1[32][H_DIM + 8];       // 25 KB
  __shared__ __attribute__((aligned(16))) _Float16 sh_fused[32][2*L_DIM + 8];  // ~17 KB ([z|mem])
  __shared__ float sh_logits[32][K_SLOTS];                                     // 4 KB
  __shared__ float sh_red[32][64];                                             // 8 KB
  __shared__ float sh_attn[32][2];
  __shared__ int   sh_idx[32][2];

  const int tid   = threadIdx.x;
  const int lane  = tid & 31;
  const int wave  = tid >> 5;
  const int mt    = wave >> 2;          // row-tile 0..1 (16 rows each)
  const int ng    = wave & 3;           // N-group 0..3
  const int lrow  = lane & 15;          // A-fragment row within tile
  const int hi8   = (lane >> 4) << 3;   // K sub-offset for upper half-wave
  const int wgrow = blockIdx.x * 32;

  // ---------------- GEMM1: h1 = gelu(x @ w1 + b1), [32 x 384] ---------------
  {
    const float* xrow = x + (size_t)(wgrow + mt * 16 + lrow) * IN_DIM;
    v8f acc[6] = {};
    for (int ks = 0; ks < 32; ++ks) {
      const int kb = (ks << 5) + hi8;
      v4f f0 = *(const v4f*)(xrow + kb);
      v4f f1 = *(const v4f*)(xrow + kb + 4);
      v4f f2 = *(const v4f*)(xrow + kb + 16);
      v4f f3 = *(const v4f*)(xrow + kb + 20);
      v16h A;
#pragma unroll
      for (int i = 0; i < 4; ++i) {
        A[i]      = (_Float16)f0[i];
        A[i + 4]  = (_Float16)f1[i];
        A[i + 8]  = (_Float16)f2[i];
        A[i + 12] = (_Float16)f3[i];
      }
#pragma unroll
      for (int t = 0; t < 6; ++t) {
        const int nt = ng * 6 + t;
        v16h Bf = *(const v16h*)(w1p + ((((nt << 5) + ks) << 9) + (lane << 4)));
        acc[t] = WMMA(A, Bf, acc[t]);
      }
    }
#pragma unroll
    for (int t = 0; t < 6; ++t) {
      const int col = (ng * 6 + t) * 16 + lrow;
      const float bb = b1[col];
#pragma unroll
      for (int j = 0; j < 8; ++j)
        sh_h1[mt * 16 + j + hi8][col] = (_Float16)gelu_exact(acc[t][j] + bb);
    }
  }
  __syncthreads();

  // ---------------- GEMM2: z = gelu(h1 @ w2 + b2), [32 x 128] ---------------
  {
    const _Float16* h1row = &sh_h1[mt * 16 + lrow][0];
    v8f acc[2] = {};
    for (int ks = 0; ks < 12; ++ks) {
      const int kb = (ks << 5) + hi8;
      v8h alo = *(const v8h*)(h1row + kb);
      v8h ahi = *(const v8h*)(h1row + kb + 16);
      v16h A = __builtin_shufflevector(alo, ahi, 0,1,2,3,4,5,6,7,8,9,10,11,12,13,14,15);
#pragma unroll
      for (int t = 0; t < 2; ++t) {
        const int nt = ng * 2 + t;
        v16h Bf = *(const v16h*)(w2p + (((nt * 12 + ks) << 9) + (lane << 4)));
        acc[t] = WMMA(A, Bf, acc[t]);
      }
    }
#pragma unroll
    for (int t = 0; t < 2; ++t) {
      const int col = (ng * 2 + t) * 16 + lrow;
      const float bb = b2[col];
#pragma unroll
      for (int j = 0; j < 8; ++j)
        sh_fused[mt * 16 + j + hi8][col] = (_Float16)gelu_exact(acc[t][j] + bb);
    }
  }
  __syncthreads();

  // ---------------- logits = (z @ mk^T) / tau, [32 x 32] ---------------------
  if (ng < 2) {                               // wave-uniform branch: EXEC all-ones
    const _Float16* zrow = &sh_fused[mt * 16 + lrow][0];
    v8f acc = {};
    for (int ks = 0; ks < 4; ++ks) {
      const int kb = (ks << 5) + hi8;
      v8h alo = *(const v8h*)(zrow + kb);
      v8h ahi = *(const v8h*)(zrow + kb + 16);
      v16h A = __builtin_shufflevector(alo, ahi, 0,1,2,3,4,5,6,7,8,9,10,11,12,13,14,15);
      v16h Bf = *(const v16h*)(mkp + ((((ng << 2) + ks) << 9) + (lane << 4)));
      acc = WMMA(A, Bf, acc);
    }
#pragma unroll
    for (int j = 0; j < 8; ++j)
      sh_logits[mt * 16 + j + hi8][(ng << 4) + lrow] = acc[j] * (1.0f / TAU);
  }
  __syncthreads();

  // ------------- top-2 over 32 slots + 2-way softmax (per row) --------------
  if (tid < 32) {
    float v0 = -3.4e38f, v1 = -3.4e38f; int i0 = 0, i1 = 0;
    for (int c = 0; c < K_SLOTS; ++c) {
      float v = sh_logits[tid][c];
      if (v > v0)      { v1 = v0; i1 = i0; v0 = v; i0 = c; }
      else if (v > v1) { v1 = v;  i1 = c; }
    }
    float e   = __expf(v1 - v0);
    float inv = 1.0f / (1.0f + e);
    sh_attn[tid][0] = inv;       // weight of top-1
    sh_attn[tid][1] = e * inv;   // weight of top-2
    sh_idx[tid][0]  = i0;
    sh_idx[tid][1]  = i1;
  }
  __syncthreads();

  // mem = a0*mv[i0] + a1*mv[i1]  -> fused[:, 128:256]
  {
    const int r  = tid >> 3;
    const int c0 = (tid & 7) << 4;
    const float a0 = sh_attn[r][0], a1 = sh_attn[r][1];
    const float* mv0 = mv + sh_idx[r][0] * L_DIM;
    const float* mv1 = mv + sh_idx[r][1] * L_DIM;
#pragma unroll
    for (int c = 0; c < 16; ++c)
      sh_fused[r][L_DIM + c0 + c] = (_Float16)(a0 * mv0[c0 + c] + a1 * mv1[c0 + c]);
  }
  __syncthreads();

  // ------- GEMM3 + head: logit = gelu(fused @ w3 + b3) @ w4 + b4 ------------
  {
    const _Float16* frow = &sh_fused[mt * 16 + lrow][0];
    v8f acc[6] = {};
    for (int ks = 0; ks < 8; ++ks) {
      const int kb = (ks << 5) + hi8;
      v8h alo = *(const v8h*)(frow + kb);
      v8h ahi = *(const v8h*)(frow + kb + 16);
      v16h A = __builtin_shufflevector(alo, ahi, 0,1,2,3,4,5,6,7,8,9,10,11,12,13,14,15);
#pragma unroll
      for (int t = 0; t < 6; ++t) {
        const int nt = ng * 6 + t;
        v16h Bf = *(const v16h*)(w3p + ((((nt << 3) + ks) << 9) + (lane << 4)));
        acc[t] = WMMA(A, Bf, acc[t]);
      }
    }
    float part[8] = {0.f,0.f,0.f,0.f,0.f,0.f,0.f,0.f};
#pragma unroll
    for (int t = 0; t < 6; ++t) {
      const int col = (ng * 6 + t) * 16 + lrow;
      const float bb = b3[col];
      const float wv = w4[col];
#pragma unroll
      for (int j = 0; j < 8; ++j)
        part[j] += gelu_exact(acc[t][j] + bb) * wv;
    }
#pragma unroll
    for (int j = 0; j < 8; ++j)
      sh_red[mt * 16 + j + hi8][(ng << 4) + lrow] = part[j];
  }
  __syncthreads();

  if (tid < 32) {
    float s = b4[0];
#pragma unroll
    for (int c = 0; c < 64; ++c) s += sh_red[tid][c];
    out[wgrow + tid] = 1.0f / (1.0f + __expf(-s));
  }
}

extern "C" void kernel_launch(void* const* d_in, const int* in_sizes, int n_in,
                              void* d_out, int out_size, void* d_ws, size_t ws_size,
                              hipStream_t stream) {
  (void)in_sizes; (void)n_in; (void)out_size; (void)ws_size;
  const float* x  = (const float*)d_in[0];
  const float* w1 = (const float*)d_in[1];
  const float* b1 = (const float*)d_in[2];
  const float* w2 = (const float*)d_in[3];
  const float* b2 = (const float*)d_in[4];
  const float* mk = (const float*)d_in[5];
  const float* mv = (const float*)d_in[6];
  const float* w3 = (const float*)d_in[7];
  const float* b3 = (const float*)d_in[8];
  const float* w4 = (const float*)d_in[9];
  const float* b4 = (const float*)d_in[10];
  float* out = (float*)d_out;
  _Float16* ws = (_Float16*)d_ws;

  // Re-pack weights into WMMA B-fragment layout every call (deterministic, ~1 MB).
  pack_b_kernel<<<(W1P_N + 255) / 256, 256, 0, stream>>>(w1, ws + W1P_OFF, IN_DIM,  H_DIM,  H_DIM, 0);
  pack_b_kernel<<<(W2P_N + 255) / 256, 256, 0, stream>>>(w2, ws + W2P_OFF, H_DIM,   L_DIM,  L_DIM, 0);
  pack_b_kernel<<<(W3P_N + 255) / 256, 256, 0, stream>>>(w3, ws + W3P_OFF, 2*L_DIM, H_DIM,  H_DIM, 0);
  pack_b_kernel<<<(MKP_N + 255) / 256, 256, 0, stream>>>(mk, ws + MKP_OFF, L_DIM,   K_SLOTS, L_DIM, 1);

  fused_forward<<<B_ROWS / 32, 256, 0, stream>>>(
      x, ws + W1P_OFF, ws + W2P_OFF, ws + W3P_OFF, ws + MKP_OFF,
      b1, b2, b3, b4, mv, w4, out);
}
